// MultiHeadAttention_33277406609799
// MI455X (gfx1250) — matrix operand order repaired and checked by reading, not compiled, not run on hardware
//
#include <hip/hip_runtime.h>

typedef __attribute__((ext_vector_type(16))) __bf16 v16bf;
typedef __attribute__((ext_vector_type(8)))  __bf16 bf16x8;
typedef __attribute__((ext_vector_type(4)))  __bf16 bf16x4;
typedef __attribute__((ext_vector_type(8)))  float  v8f;

// ---------------------------------------------------------------------------
// fragment helpers (CDNA5 ISA layouts, wave32)
// ---------------------------------------------------------------------------
__device__ __forceinline__ v16bf make_frag16(const __bf16* p0, const __bf16* p1) {
  bf16x8 lo = *(const bf16x8*)p0;
  bf16x8 hi = *(const bf16x8*)p1;
  v16bf f;
#pragma unroll
  for (int i = 0; i < 8; ++i) { f[i] = lo[i]; f[i + 8] = hi[i]; }
  return f;
}
// A-fragment 16x32: lane<16 holds K {0..7,16..23}; lane>=16 holds K {8..15,24..31}
__device__ __forceinline__ v16bf frag_a(const __bf16* row, int kb /*0|8*/) {
  return make_frag16(row + kb, row + kb + 16);
}
// B-fragment 32x16: lane<16 holds K {0..7,16..23}; lane>=16 holds K {8..15,24..31}
__device__ __forceinline__ v16bf frag_b(const __bf16* row, int kbB /*0|8*/) {
  return make_frag16(row + kbB, row + kbB + 16);
}

// async 16B global -> LDS copy (ASYNCcnt-tracked, gfx1250)
__device__ __forceinline__ void async_cp16(unsigned lds_off, const void* gaddr) {
  asm volatile("global_load_async_to_lds_b128 %0, %1, off"
               :: "v"(lds_off), "v"(gaddr) : "memory");
}
__device__ __forceinline__ void wait_async0() {
  asm volatile("s_wait_asynccnt 0x0" ::: "memory");
}

// max across lanes within 16-lane halves via immediate ds_swizzle (xor 1,2,4,8)
__device__ __forceinline__ float xor16_max(float v) {
  union { float f; int i; } u, w;
  u.f = v;
  w.i = __builtin_amdgcn_ds_swizzle(u.i, 0x041F); u.f = fmaxf(u.f, w.f);
  w.i = __builtin_amdgcn_ds_swizzle(u.i, 0x081F); u.f = fmaxf(u.f, w.f);
  w.i = __builtin_amdgcn_ds_swizzle(u.i, 0x101F); u.f = fmaxf(u.f, w.f);
  w.i = __builtin_amdgcn_ds_swizzle(u.i, 0x201F); u.f = fmaxf(u.f, w.f);
  return u.f;
}

// ---------------------------------------------------------------------------
// fp32 -> bf16 convert (vectorized x4)
// ---------------------------------------------------------------------------
__global__ __launch_bounds__(256) void cvt_f32_bf16(const float* __restrict__ in,
                                                    __bf16* __restrict__ out, int n4) {
  int i = blockIdx.x * 256 + threadIdx.x;
  if (i < n4) {
    const float4 v = ((const float4*)in)[i];
    bf16x4 o;
    o[0] = (__bf16)v.x; o[1] = (__bf16)v.y; o[2] = (__bf16)v.z; o[3] = (__bf16)v.w;
    ((bf16x4*)out)[i] = o;
  }
}

// ---------------------------------------------------------------------------
// NT GEMM: C[m,n] = sum_k A[m,k] * W[n,k], async double-buffered LDS staging,
// one barrier per pipelined k-step.
// MODE 0: bf16 [M][N] | MODE 1: bf16 per-head transposed [n][h][d][s] | MODE 2: f32+bias
// ---------------------------------------------------------------------------
#define BM 128
#define BN 128
#define BK 32
#define LDT 40   // padded LDS row stride (bf16): 80B, 16B-aligned, conflict-free

template <int MODE>
__global__ __launch_bounds__(256) void gemm_nt(const __bf16* __restrict__ A,
                                               const __bf16* __restrict__ W,
                                               __bf16* __restrict__ Cb,
                                               float* __restrict__ Cf,
                                               const float* __restrict__ bias,
                                               int M, int Ncols, int K) {
  __shared__ __bf16 As[2][BM * LDT];
  __shared__ __bf16 Bs[2][BN * LDT];
  const int t    = threadIdx.x;
  const int lane = t & 31;
  const int wave = t >> 5;
  const int m0 = blockIdx.y * BM;
  const int n0 = blockIdx.x * BN;
  const int tm0 = (wave >> 1) * 32;
  const int tn0 = (wave & 1) * 64;
  const int col = lane & 15;
  const int kbA = (lane & 16) ? 8 : 0;
  const int kbB = (lane & 16) ? 8 : 0;

  v8f acc[2][4];
#pragma unroll
  for (int i = 0; i < 2; ++i)
#pragma unroll
    for (int j = 0; j < 4; ++j)
#pragma unroll
      for (int r = 0; r < 8; ++r) acc[i][j][r] = 0.0f;

  const int srow = t >> 1;
  const int shalf = (t & 1) * 16;
  unsigned dA[2], dB[2];
#pragma unroll
  for (int p = 0; p < 2; ++p) {
    dA[p] = (unsigned)(uintptr_t)&As[p][srow * LDT + shalf];
    dB[p] = (unsigned)(uintptr_t)&Bs[p][srow * LDT + shalf];
  }
  auto stage = [&](int k0, int p) {
    const __bf16* pa = A + (size_t)(m0 + srow) * K + k0 + shalf;
    const __bf16* pb = W + (size_t)(n0 + srow) * K + k0 + shalf;
    async_cp16(dA[p], pa);      async_cp16(dA[p] + 16, pa + 8);
    async_cp16(dB[p], pb);      async_cp16(dB[p] + 16, pb + 8);
  };

  const int nk = K / BK;
  stage(0, 0);
  for (int kc = 0; kc < nk; ++kc) {
    wait_async0();
    __syncthreads();                       // single barrier per k-step
    if (kc + 1 < nk) stage((kc + 1) * BK, (kc + 1) & 1);
    const int p = kc & 1;

    v16bf af[2], bf[4];
#pragma unroll
    for (int i = 0; i < 2; ++i) af[i] = frag_a(&As[p][(tm0 + i * 16 + col) * LDT], kbA);
#pragma unroll
    for (int j = 0; j < 4; ++j) bf[j] = frag_b(&Bs[p][(tn0 + j * 16 + col) * LDT], kbB);
#pragma unroll
    for (int i = 0; i < 2; ++i)
#pragma unroll
      for (int j = 0; j < 4; ++j)
        acc[i][j] = __builtin_amdgcn_wmma_f32_16x16x32_bf16(
            false, af[i], false, bf[j], (short)0, acc[i][j], false, false);
  }

  const int rb = (lane & 16) ? 8 : 0;
#pragma unroll
  for (int i = 0; i < 2; ++i)
#pragma unroll
    for (int j = 0; j < 4; ++j)
#pragma unroll
      for (int r = 0; r < 8; ++r) {
        const int gm = m0 + tm0 + i * 16 + rb + r;
        const int gn = n0 + tn0 + j * 16 + col;
        const float v = acc[i][j][r];
        if (MODE == 0) {
          Cb[(size_t)gm * Ncols + gn] = (__bf16)v;
        } else if (MODE == 1) {
          const int h = gn >> 6, d = gn & 63;
          const int nb = gm >> 11, s = gm & 2047;
          Cb[(((size_t)nb * 16 + h) * 64 + d) * 2048 + s] = (__bf16)v;
        } else {
          Cf[(size_t)gm * Ncols + gn] = v + bias[gn];
        }
      }
}

// ---------------------------------------------------------------------------
// Flash attention (causal), KC=64 chunks: 18 WMMA per chunk (8 QK^T + 2 row-sum
// + 8 PV), async double-buffered K/V staging, scalar-uniform causal branches.
// Qp/Kp: bf16 [n][s][h][d]; VpT: bf16 [n][h][d][s]; AO: bf16 [n][s][h][d].
// ---------------------------------------------------------------------------
#define BQ 128
#define KC 64
#define KLD 72   // Ks row stride (d-dim): 144B, 16B-aligned
#define VLD 72   // Vs row stride (k-dim)
#define PLD 72   // Ps row stride

__global__ __launch_bounds__(256) void flash_attn(const __bf16* __restrict__ Qp,
                                                  const __bf16* __restrict__ Kp,
                                                  const __bf16* __restrict__ VpT,
                                                  __bf16* __restrict__ AO) {
  __shared__ __bf16 Ks[2][64 * KLD];     // [k][d]
  __shared__ __bf16 Vs[2][64 * VLD];     // [d][k] (transposed V chunk)
  __shared__ __bf16 Ps[8 * 16 * PLD];    // per-wave P tile [16][64]
  const int t    = threadIdx.x;
  const int lane = t & 31;
  const int wave = t >> 5;
  const int h  = blockIdx.y;
  const int nb = blockIdx.z;
  const int q0wg = blockIdx.x * BQ;
  const int q0 = q0wg + wave * 16;
  const int q0u = __builtin_amdgcn_readfirstlane(q0);   // scalar wave-uniform copy
  const int col = lane & 15;
  const int kbA = (lane & 16) ? 8 : 0;
  const int kbB = (lane & 16) ? 8 : 0;
  const int rb  = (lane & 16) ? 8 : 0;

  // resident Q A-fragments (2 d-steps)
  v16bf qf[2];
  {
    const __bf16* qrow = Qp + (((size_t)nb * 2048 + q0 + col) * 16 + h) * 64;
#pragma unroll
    for (int ds = 0; ds < 2; ++ds) qf[ds] = frag_a(qrow + ds * 32, kbA);
  }
  v16bf onesf;
#pragma unroll
  for (int i = 0; i < 16; ++i) onesf[i] = (__bf16)1.0f;

  v8f o[4];
#pragma unroll
  for (int tt = 0; tt < 4; ++tt)
#pragma unroll
    for (int r = 0; r < 8; ++r) o[tt][r] = 0.0f;
  float mrow[8], lrow[8];
#pragma unroll
  for (int r = 0; r < 8; ++r) { mrow[r] = -3.0e38f; lrow[r] = 0.0f; }

  const int nchunks = (q0wg + BQ) / KC;          // workgroup-uniform causal bound

  const int srow = t >> 2;                       // 0..63 (k-row for K, d-row for V)
  const int sseg = (t & 3) * 16;                 // 16-element segment
  unsigned dK[2], dV[2];
#pragma unroll
  for (int p = 0; p < 2; ++p) {
    dK[p] = (unsigned)(uintptr_t)&Ks[p][srow * KLD + sseg];
    dV[p] = (unsigned)(uintptr_t)&Vs[p][srow * VLD + sseg];
  }
  auto stage = [&](int c, int p) {
    const int k0 = c * KC;
    const __bf16* kp = Kp + (((size_t)nb * 2048 + k0 + srow) * 16 + h) * 64 + sseg;
    async_cp16(dK[p], kp);  async_cp16(dK[p] + 16, kp + 8);
    const __bf16* vp = VpT + (((size_t)nb * 16 + h) * 64 + srow) * 2048 + k0 + sseg;
    async_cp16(dV[p], vp);  async_cp16(dV[p] + 16, vp + 8);
  };

  stage(0, 0);
  for (int c = 0; c < nchunks; ++c) {
    wait_async0();
    __syncthreads();                             // single barrier per chunk
    if (c + 1 < nchunks) stage(c + 1, (c + 1) & 1);
    const int p = c & 1;
    const int k0 = c * KC;

    if (k0 <= q0u + 15) {                        // scalar branch, EXEC stays full
      // S = Q K^T (8 WMMA): 4 n-tiles x 2 d-steps
      v8f s[4];
#pragma unroll
      for (int j = 0; j < 4; ++j) {
        v8f z;
#pragma unroll
        for (int r = 0; r < 8; ++r) z[r] = 0.0f;
#pragma unroll
        for (int ds = 0; ds < 2; ++ds) {
          v16bf kf = frag_b(&Ks[p][(j * 16 + col) * KLD + ds * 32], kbB);
          z = __builtin_amdgcn_wmma_f32_16x16x32_bf16(false, qf[ds], false, kf,
                                                      (short)0, z, false, false);
        }
        s[j] = z;
      }
      // scale; mask only on the diagonal chunk (scalar-uniform branch)
      float e[4][8];
      if (k0 + KC - 1 > q0u) {
#pragma unroll
        for (int j = 0; j < 4; ++j)
#pragma unroll
          for (int r = 0; r < 8; ++r) {
            const int kk = k0 + j * 16 + col;
            const int qq = q0 + rb + r;
            e[j][r] = (kk <= qq) ? s[j][r] * 0.125f : -3.0e38f;
          }
      } else {
#pragma unroll
        for (int j = 0; j < 4; ++j)
#pragma unroll
          for (int r = 0; r < 8; ++r) e[j][r] = s[j][r] * 0.125f;
      }
      // online softmax: row max via ds_swizzle, rescale, exp -> P (bf16 in LDS)
#pragma unroll
      for (int r = 0; r < 8; ++r) {
        const float rm = xor16_max(fmaxf(fmaxf(e[0][r], e[1][r]),
                                         fmaxf(e[2][r], e[3][r])));
        const float mn = fmaxf(mrow[r], rm);
        const float alpha = __expf(mrow[r] - mn);
        mrow[r] = mn;
        lrow[r] *= alpha;
#pragma unroll
        for (int tt = 0; tt < 4; ++tt) o[tt][r] *= alpha;
        __bf16* prow = &Ps[((size_t)wave * 16 + rb + r) * PLD];
#pragma unroll
        for (int j = 0; j < 4; ++j)
          prow[j * 16 + col] = (__bf16)__expf(e[j][r] - mn);
      }
      asm volatile("s_wait_dscnt 0x0" ::: "memory");   // same-wave LDS RAW
      const __bf16* prow = &Ps[((size_t)wave * 16 + col) * PLD];
      v16bf pf0 = frag_a(prow, kbA);
      v16bf pf1 = frag_a(prow + 32, kbA);
      // row sums on the matrix pipe (2 WMMA): ls[r] = sum_k P[r,k]
      {
        v8f z;
#pragma unroll
        for (int r = 0; r < 8; ++r) z[r] = 0.0f;
        v8f ls = __builtin_amdgcn_wmma_f32_16x16x32_bf16(false, pf0, false, onesf,
                                                         (short)0, z, false, false);
        ls = __builtin_amdgcn_wmma_f32_16x16x32_bf16(false, pf1, false, onesf,
                                                     (short)0, ls, false, false);
#pragma unroll
        for (int r = 0; r < 8; ++r) lrow[r] += ls[r];
      }
      // O += P V (8 WMMA): 4 d-tiles x 2 k-steps
#pragma unroll
      for (int tt = 0; tt < 4; ++tt) {
        v16bf vf0 = frag_b(&Vs[p][(tt * 16 + col) * VLD], kbB);
        o[tt] = __builtin_amdgcn_wmma_f32_16x16x32_bf16(false, pf0, false, vf0,
                                                        (short)0, o[tt], false, false);
        v16bf vf1 = frag_b(&Vs[p][(tt * 16 + col) * VLD + 32], kbB);
        o[tt] = __builtin_amdgcn_wmma_f32_16x16x32_bf16(false, pf1, false, vf1,
                                                        (short)0, o[tt], false, false);
      }
    }
  }

#pragma unroll
  for (int r = 0; r < 8; ++r) {
    const float inv = 1.0f / lrow[r];
    const int qq = q0 + rb + r;
#pragma unroll
    for (int tt = 0; tt < 4; ++tt)
      AO[(((size_t)nb * 2048 + qq) * 16 + h) * 64 + tt * 16 + col] =
          (__bf16)(o[tt][r] * inv);
  }
}

// ---------------------------------------------------------------------------
// launch
// ---------------------------------------------------------------------------
extern "C" void kernel_launch(void* const* d_in, const int* in_sizes, int n_in,
                              void* d_out, int out_size, void* d_ws, size_t ws_size,
                              hipStream_t stream) {
  const float* values = (const float*)d_in[0];
  const float* keys   = (const float*)d_in[1];
  const float* query  = (const float*)d_in[2];
  // d_in[3] = mask (causal, derived analytically in-kernel)
  const float* Wv = (const float*)d_in[4];
  const float* Wk = (const float*)d_in[5];
  const float* Wq = (const float*)d_in[6];
  const float* Wo = (const float*)d_in[7];
  const float* bo = (const float*)d_in[8];
  float* out = (float*)d_out;

  const int M = 8192, E = 1024, K = 1024;
  const size_t SZX = (size_t)M * E * sizeof(__bf16);
  const size_t SZW = (size_t)E * E * sizeof(__bf16);
  char* ws = (char*)d_ws;
  __bf16* Xv  = (__bf16*)(ws);
  __bf16* Xk  = (__bf16*)(ws + SZX);
  __bf16* Xq  = (__bf16*)(ws + 2 * SZX);
  __bf16* Wvb = (__bf16*)(ws + 3 * SZX);
  __bf16* Wkb = (__bf16*)(ws + 3 * SZX + SZW);
  __bf16* Wqb = (__bf16*)(ws + 3 * SZX + 2 * SZW);
  __bf16* Wob = (__bf16*)(ws + 3 * SZX + 3 * SZW);
  __bf16* VpT = (__bf16*)(ws + 3 * SZX + 4 * SZW);
  __bf16* Kp  = (__bf16*)(ws + 4 * SZX + 4 * SZW);
  __bf16* Qp  = (__bf16*)(ws + 5 * SZX + 4 * SZW);
  __bf16* AO  = (__bf16*)(ws + 6 * SZX + 4 * SZW);

  const int nX4 = M * E / 4, nW4 = E * E / 4;
  cvt_f32_bf16<<<nX4 / 256, 256, 0, stream>>>(values, Xv, nX4);
  cvt_f32_bf16<<<nX4 / 256, 256, 0, stream>>>(keys,   Xk, nX4);
  cvt_f32_bf16<<<nX4 / 256, 256, 0, stream>>>(query,  Xq, nX4);
  cvt_f32_bf16<<<nW4 / 256, 256, 0, stream>>>(Wv, Wvb, nW4);
  cvt_f32_bf16<<<nW4 / 256, 256, 0, stream>>>(Wk, Wkb, nW4);
  cvt_f32_bf16<<<nW4 / 256, 256, 0, stream>>>(Wq, Wqb, nW4);
  cvt_f32_bf16<<<nW4 / 256, 256, 0, stream>>>(Wo, Wob, nW4);

  dim3 gg(E / BN, M / BM);
  gemm_nt<1><<<gg, 256, 0, stream>>>(Xv, Wvb, VpT, nullptr, nullptr, M, E, K);
  gemm_nt<0><<<gg, 256, 0, stream>>>(Xk, Wkb, Kp,  nullptr, nullptr, M, E, K);
  gemm_nt<0><<<gg, 256, 0, stream>>>(Xq, Wqb, Qp,  nullptr, nullptr, M, E, K);

  flash_attn<<<dim3(2048 / BQ, 16, 4), 256, 0, stream>>>(Qp, Kp, VpT, AO);

  gemm_nt<2><<<gg, 256, 0, stream>>>(AO, Wob, nullptr, out, bo, M, E, K);
}
